// BiLSTM_CRF_84310208020831
// MI455X (gfx1250) — matrix operand order repaired and checked by reading, not compiled, not run on hardware
//
#include <hip/hip_runtime.h>
#include <hip/hip_bf16.h>

// ---------------- problem constants ----------------
constexpr int TT   = 4096;   // sequence length
constexpr int EMB  = 300;
constexpr int DRAW = 1200;   // 4*EMB
constexpr int DP   = 1216;   // padded to multiple of 32 (38 K-steps)
constexpr int HID  = 256;
constexpr int G4   = 1024;   // 4*HID
constexpr int KTAG = 12;
constexpr int START_TAG = 10;
constexpr int STOP_TAG  = 11;
constexpr float NEGV = -10000.0f;

// ---------------- WMMA vector types ----------------
typedef __attribute__((ext_vector_type(16))) _Float16 v16h;
typedef __attribute__((ext_vector_type(8)))  _Float16 v8h;
typedef __attribute__((ext_vector_type(8)))  float    v8f;

__device__ inline v16h cat16(v8h lo, v8h hi) {
  return __builtin_shufflevector(lo, hi, 0,1,2,3,4,5,6,7,8,9,10,11,12,13,14,15);
}
__device__ inline v8f wmma16(v16h a, v16h b, v8f c) {
  return __builtin_amdgcn_wmma_f32_16x16x32_f16(false, a, false, b, (short)0, c, false, false);
}
__device__ inline float sigm(float x) { return 1.0f / (1.0f + __expf(-x)); }

// ---------------- workspace layout (bytes) ----------------
constexpr size_t OFF_EMB   = 0;                                   // T*DP f16
constexpr size_t SZ_EMB    = (size_t)TT * DP * 2;
constexpr size_t OFF_WIHT  = OFF_EMB + SZ_EMB;                    // 2*DP*G4 f16  [n][k]
constexpr size_t SZ_WIHT   = (size_t)2 * DP * G4 * 2;
constexpr size_t OFF_WHHT  = OFF_WIHT + SZ_WIHT;                  // 2*HID*G4 f16 [n][k]
constexpr size_t SZ_WHHT   = (size_t)2 * HID * G4 * 2;
constexpr size_t OFF_WTAGT = OFF_WHHT + SZ_WHHT;                  // 512*16 f16   [n][k]
constexpr size_t SZ_WTAGT  = (size_t)512 * 16 * 2;
constexpr size_t OFF_XG    = OFF_WTAGT + SZ_WTAGT;                // 2*T*G4 f32
constexpr size_t SZ_XG     = (size_t)2 * TT * G4 * 4;
constexpr size_t OFF_HSH   = OFF_XG + SZ_XG;                      // 2*T*H f16
constexpr size_t SZ_HSH    = (size_t)2 * TT * HID * 2;
constexpr size_t OFF_FEATS = OFF_HSH + SZ_HSH;                    // T*12 f32
constexpr size_t SZ_FEATS  = (size_t)TT * KTAG * 4;
constexpr size_t OFF_BP    = OFF_FEATS + SZ_FEATS;                // T*12 i32

// =====================================================================
// K0a: gather 4 embeddings per timestep into padded f16 matrix (T x DP)
// =====================================================================
__global__ __launch_bounds__(256) void embed_kernel(
    const int* __restrict__ w0, const int* __restrict__ w1,
    const int* __restrict__ w2, const int* __restrict__ w3,
    const float* __restrict__ W_emb, _Float16* __restrict__ emb_h)
{
  int idx = blockIdx.x * 256 + threadIdx.x;          // < TT*DP
  int t = idx / DP;
  int d = idx - t * DP;
  float v = 0.0f;
  if (d < DRAW) {
    int seg = d / EMB;
    int dd  = d - seg * EMB;
    int word = (seg == 0) ? w0[t] : (seg == 1) ? w1[t] : (seg == 2) ? w2[t] : w3[t];
    v = W_emb[(size_t)word * EMB + dd];
  }
  emb_h[idx] = (_Float16)v;
}

// =====================================================================
// K0b: pad+convert w_ih -> [dir][n][k] f16 (B-fragment layout:
// lane = N, elements = K, so per-lane loads become contiguous)
// =====================================================================
__global__ __launch_bounds__(256) void wihT_kernel(
    const float* __restrict__ wf, const float* __restrict__ wb,
    _Float16* __restrict__ wihT)
{
  int idx = blockIdx.x * 256 + threadIdx.x;          // < 2*DP*G4
  int dir = idx / (DP * G4);
  int rem = idx - dir * (DP * G4);
  int n = rem / DP;
  int k = rem - n * DP;
  const float* w = dir ? wb : wf;
  float v = (k < DRAW) ? w[(size_t)n * DRAW + k] : 0.0f;
  wihT[idx] = (_Float16)v;
}

// K0c: convert w_hh -> [dir][n][k] f16
__global__ __launch_bounds__(256) void whhT_kernel(
    const float* __restrict__ wf, const float* __restrict__ wb,
    _Float16* __restrict__ whhT)
{
  int idx = blockIdx.x * 256 + threadIdx.x;          // < 2*HID*G4
  int dir = idx / (HID * G4);
  int rem = idx - dir * (HID * G4);
  int n = rem / HID;
  int k = rem - n * HID;
  const float* w = dir ? wb : wf;
  whhT[idx] = (_Float16)w[(size_t)n * HID + k];
}

// K0d: convert+pad W_tag -> [n16][k] f16
__global__ __launch_bounds__(256) void wtagT_kernel(
    const float* __restrict__ Wtag, _Float16* __restrict__ wtagT)
{
  int idx = blockIdx.x * 256 + threadIdx.x;          // < 512*16
  int n = idx >> 9;
  int k = idx & 511;
  float v = (n < KTAG) ? Wtag[(size_t)n * (2 * HID) + k] : 0.0f;
  wtagT[idx] = (_Float16)v;
}

// =====================================================================
// K1: xg[dir] = emb(_rev) @ w_ih^T + (b_ih + b_hh)     (WMMA f16->f32)
// Block = 8 waves sharing one M-tile (A panel staged in LDS via
// global_load_async_to_lds_b128); each wave owns 4 N-tiles; B stream is
// ping-pong double-buffered (unroll-by-2, no register rotation moves).
// =====================================================================
__global__ __launch_bounds__(256) void xg_gemm_kernel(
    const _Float16* __restrict__ emb_h, const _Float16* __restrict__ wihT,
    const float* __restrict__ bihf, const float* __restrict__ bhhf,
    const float* __restrict__ bihb, const float* __restrict__ bhhb,
    float* __restrict__ xg)
{
  __shared__ __align__(16) _Float16 apan[16 * DP];   // 38 KB A panel
  int tid = threadIdx.x, lane = tid & 31, wave = tid >> 5;
  int bid = blockIdx.x;
  int dir   = bid >> 9;                 // 0..1
  int rem   = bid & 511;
  int mtile = rem >> 1;                 // 0..255
  int nhalf = rem & 1;
  int mbase = mtile << 4;
  int nbase = nhalf * 512 + wave * 64;  // 4 N-tiles per wave

  // ---- async-stage A panel (16 rows x DP halves) into LDS ----
  constexpr int CHUNKS = 16 * DP / 8;   // 16-byte chunks = 2432
  for (int c = tid; c < CHUNKS; c += 256) {
    int row = c / (DP / 8);
    int col = c - row * (DP / 8);
    int trow = mbase + row;
    int erow = dir ? (TT - 1 - trow) : trow;  // backward dir reads reversed time
    const _Float16* gp = emb_h + (size_t)erow * DP + col * 8;
    unsigned ldsoff = (unsigned)(row * DP + col * 8) * 2u;  // panel at LDS base 0
    asm volatile("global_load_async_to_lds_b128 %0, %1, off"
                 :: "v"(ldsoff), "v"((unsigned long long)(uintptr_t)gp)
                 : "memory");
  }
  asm volatile("s_wait_asynccnt 0" ::: "memory");
  __syncthreads();

  int r = lane & 15, khalf = (lane >> 4) << 3;   // A: lanes split K by 8
  const _Float16* wT = wihT + (size_t)dir * DP * G4;

  v8f acc[4];
#pragma unroll
  for (int j = 0; j < 4; ++j) acc[j] = (v8f){};

  auto loadA = [&](int ks) -> v16h {
    int base = r * DP + ks * 32 + khalf;
    v8h lo = *(const v8h*)(apan + base);
    v8h hi = *(const v8h*)(apan + base + 16);
    return cat16(lo, hi);
  };
  // B: lane = N, elements = K in table order (contiguous in wihT [n][k])
  auto loadB = [&](int ks, v16h (&bf)[4]) {
    int k = ks * 32 + khalf;
    const _Float16* rowp = wT + (size_t)(nbase + r) * DP + k;
#pragma unroll
    for (int j = 0; j < 4; ++j) {
      const _Float16* p = rowp + (size_t)j * 16 * DP;
      bf[j] = cat16(*(const v8h*)p, *(const v8h*)(p + 16));
    }
  };

  v16h a0 = loadA(0), a1;
  v16h b0[4], b1[4];
  loadB(0, b0);

  // ping-pong, unroll-by-2: even step lives in (a0,b0), odd in (a1,b1)
  for (int ks = 0; ks < DP / 32 - 2; ks += 2) {
    a1 = loadA(ks + 1); loadB(ks + 1, b1);
#pragma unroll
    for (int j = 0; j < 4; ++j) acc[j] = wmma16(a0, b0[j], acc[j]);
    a0 = loadA(ks + 2); loadB(ks + 2, b0);
#pragma unroll
    for (int j = 0; j < 4; ++j) acc[j] = wmma16(a1, b1[j], acc[j]);
  }
  // epilogue: steps 36 (in buf0) and 37
  a1 = loadA(DP / 32 - 1); loadB(DP / 32 - 1, b1);
#pragma unroll
  for (int j = 0; j < 4; ++j) acc[j] = wmma16(a0, b0[j], acc[j]);
#pragma unroll
  for (int j = 0; j < 4; ++j) acc[j] = wmma16(a1, b1[j], acc[j]);

  const float* bih = dir ? bihb : bihf;
  const float* bhh = dir ? bhhb : bhhf;
  float* xgo = xg + (size_t)dir * TT * G4;
  int rowadd = (lane >> 4) << 3;                            // 0 or 8
#pragma unroll
  for (int j = 0; j < 4; ++j) {
    int N = nbase + j * 16 + r;
    float bs = bih[N] + bhh[N];
#pragma unroll
    for (int vr = 0; vr < 8; ++vr)
      xgo[(size_t)(mbase + vr + rowadd) * G4 + N] = acc[j][vr] + bs;
  }
}

// =====================================================================
// K2: sequential LSTM recurrence, one 1024-thread block (32 waves) per
// direction. w_hh (transposed) resident in VGPR B-fragments; h broadcast
// via LDS as f16; xg_t seeded into accumulator row 0. K-reduction split
// into two independent 4-deep WMMA chains to halve per-step latency.
// =====================================================================
__global__ __launch_bounds__(1024) void lstm_rec_kernel(
    const float* __restrict__ xg,
    const _Float16* __restrict__ whhT,
    _Float16* __restrict__ hs_h)
{
  __shared__ float gates[G4];
  __shared__ __align__(16) _Float16 h_sh[HID + 32];
  int dir = blockIdx.x;
  const _Float16* wT = whhT + (size_t)dir * HID * G4;
  const float* xgd = xg + (size_t)dir * TT * G4;
  _Float16* hsd = hs_h + (size_t)dir * TT * HID;

  int tid = threadIdx.x, lane = tid & 31, wave = tid >> 5;
  int r = lane & 15, khalf = (lane >> 4) << 3;
  int nt0 = wave * 2, nt1 = nt0 + 1;

  // resident B-fragments: lane = N, elements = K in table order (contiguous)
  v16h bfr[2][8];
#pragma unroll
  for (int jt = 0; jt < 2; ++jt) {
#pragma unroll
    for (int kt = 0; kt < 8; ++kt) {
      int k = kt * 32 + khalf;
      const _Float16* p = wT + (size_t)((nt0 + jt) * 16 + r) * HID + k;
      bfr[jt][kt] = cat16(*(const v8h*)p, *(const v8h*)(p + 16));
    }
  }

  if (tid < HID + 32) h_sh[tid] = (_Float16)0;
  float cst = 0.0f;
  __syncthreads();

  bool amask = (r == 0);           // only M row 0 of the A fragment is live
  v16h az = (v16h){};

  for (int t = 0; t < TT; ++t) {
    // two independent 4-deep chains per tile; xg_t seeds chain A's row 0
    v8f acc0a = (v8f){}, acc0b = (v8f){}, acc1a = (v8f){}, acc1b = (v8f){};
    float x0 = xgd[(size_t)t * G4 + nt0 * 16 + r];
    float x1 = xgd[(size_t)t * G4 + nt1 * 16 + r];
    acc0a[0] = (lane < 16) ? x0 : 0.0f;
    acc1a[0] = (lane < 16) ? x1 : 0.0f;

    v16h afr[8];
#pragma unroll
    for (int kt = 0; kt < 8; ++kt) {
      int base = kt * 32 + khalf;
      v8h lo = *(const v8h*)(h_sh + base);
      v8h hi = *(const v8h*)(h_sh + base + 16);
      afr[kt] = amask ? cat16(lo, hi) : az;
    }
#pragma unroll
    for (int kt = 0; kt < 4; ++kt) {
      acc0a = wmma16(afr[kt],     bfr[0][kt],     acc0a);
      acc0b = wmma16(afr[kt + 4], bfr[0][kt + 4], acc0b);
      acc1a = wmma16(afr[kt],     bfr[1][kt],     acc1a);
      acc1b = wmma16(afr[kt + 4], bfr[1][kt + 4], acc1b);
    }
    if (lane < 16) {
      gates[nt0 * 16 + lane] = acc0a[0] + acc0b[0];
      gates[nt1 * 16 + lane] = acc1a[0] + acc1b[0];
    }
    __syncthreads();

    if (tid < HID) {
      float gi = sigm(gates[tid]);
      float gf = sigm(gates[HID + tid]);
      float gg = tanhf(gates[2 * HID + tid]);
      float go = sigm(gates[3 * HID + tid]);
      cst = gf * cst + gi * gg;
      _Float16 hh = (_Float16)(go * tanhf(cst));
      h_sh[tid] = hh;
      hsd[(size_t)t * HID + tid] = hh;
    }
    __syncthreads();
  }
}

// =====================================================================
// K3: feats = [hf ; hb_reversed] @ W_tag^T + b_tag   (WMMA, N padded 16)
// one wave per M-tile of 16 timesteps; A from f16 hidden states (vector
// loads), B from padded wtagT [n][k]; fully unrolled for scheduling.
// =====================================================================
__global__ __launch_bounds__(256) void feats_kernel(
    const _Float16* __restrict__ hs_h, const _Float16* __restrict__ wtagT,
    const float* __restrict__ btag, float* __restrict__ feats)
{
  int tid = threadIdx.x, lane = tid & 31, wave = tid >> 5;
  int mtile = blockIdx.x * 8 + wave;
  int mbase = mtile << 4;
  int r = lane & 15, khalf = (lane >> 4) << 3;
  int trow = mbase + r;
  const _Float16* hf = hs_h;
  const _Float16* hb = hs_h + (size_t)TT * HID;

  v8f acc = (v8f){};
#pragma unroll
  for (int ks = 0; ks < 16; ++ks) {
    int k0 = ks * 32;
    const _Float16* arow = (ks < 8)
        ? (hf + (size_t)trow * HID + k0)
        : (hb + (size_t)(TT - 1 - trow) * HID + (k0 - HID));
    v8h alo = *(const v8h*)(arow + khalf);
    v8h ahi = *(const v8h*)(arow + khalf + 16);
    v16h a = cat16(alo, ahi);
    int k = k0 + khalf;                        // B: lane = N, elements = K
    const _Float16* p = wtagT + (size_t)r * 512 + k;
    v16h b = cat16(*(const v8h*)p, *(const v8h*)(p + 16));
    acc = wmma16(a, b, acc);
  }

  if (r < KTAG) {
    float bt = btag[r];
    int rowadd = (lane >> 4) << 3;
#pragma unroll
    for (int vr = 0; vr < 8; ++vr)
      feats[(size_t)(mbase + vr + rowadd) * KTAG + r] = acc[vr] + bt;
  }
}

// =====================================================================
// K4: Viterbi decode — single wave; exact first-max argmax semantics
// =====================================================================
__global__ __launch_bounds__(32) void viterbi_kernel(
    const float* __restrict__ feats, const float* __restrict__ trans,
    int* __restrict__ bp, float* __restrict__ out)
{
  __shared__ float fv[16];
  __shared__ float term[16];
  int tid = threadIdx.x;
  if (tid < KTAG) fv[tid] = (tid == START_TAG) ? 0.0f : NEGV;
  __syncthreads();

  for (int t = 0; t < TT; ++t) {
    float nv = 0.0f; int bi = 0;
    if (tid < KTAG) {
      float m = fv[0] + trans[tid * KTAG + 0]; bi = 0;
      for (int p = 1; p < KTAG; ++p) {
        float v = fv[p] + trans[tid * KTAG + p];
        if (v > m) { m = v; bi = p; }
      }
      nv = m + feats[(size_t)t * KTAG + tid];
    }
    __syncthreads();
    if (tid < KTAG) { fv[tid] = nv; bp[(size_t)t * KTAG + tid] = bi; }
    __syncthreads();
  }

  if (tid < KTAG) {
    float v = fv[tid] + trans[STOP_TAG * KTAG + tid];
    if (tid == STOP_TAG || tid == START_TAG) v = NEGV;
    term[tid] = v;
  }
  __syncthreads();
  if (tid == 0) {
    int best = 0; float bs = term[0];
    for (int k = 1; k < KTAG; ++k) if (term[k] > bs) { bs = term[k]; best = k; }
    out[0] = bs;
    int tag = best;
    for (int t = TT - 1; t >= 0; --t) {
      out[1 + t] = (float)tag;
      tag = bp[(size_t)t * KTAG + tag];
    }
  }
}

// =====================================================================
extern "C" void kernel_launch(void* const* d_in, const int* in_sizes, int n_in,
                              void* d_out, int out_size, void* d_ws, size_t ws_size,
                              hipStream_t stream)
{
  const int*   words  = (const int*)d_in[0];
  const int*   words1 = (const int*)d_in[1];
  const int*   words2 = (const int*)d_in[2];
  const int*   words3 = (const int*)d_in[3];
  const float* W_emb  = (const float*)d_in[4];
  const float* w_ih_f = (const float*)d_in[5];
  const float* w_hh_f = (const float*)d_in[6];
  const float* b_ih_f = (const float*)d_in[7];
  const float* b_hh_f = (const float*)d_in[8];
  const float* w_ih_b = (const float*)d_in[9];
  const float* w_hh_b = (const float*)d_in[10];
  const float* b_ih_b = (const float*)d_in[11];
  const float* b_hh_b = (const float*)d_in[12];
  const float* W_tag  = (const float*)d_in[13];
  const float* b_tag  = (const float*)d_in[14];
  const float* trans  = (const float*)d_in[15];
  float* out = (float*)d_out;

  char* ws = (char*)d_ws;
  _Float16* emb_h = (_Float16*)(ws + OFF_EMB);
  _Float16* wihT  = (_Float16*)(ws + OFF_WIHT);
  _Float16* whhT  = (_Float16*)(ws + OFF_WHHT);
  _Float16* wtagT = (_Float16*)(ws + OFF_WTAGT);
  float*    xg    = (float*)(ws + OFF_XG);
  _Float16* hs_h  = (_Float16*)(ws + OFF_HSH);
  float*    feats = (float*)(ws + OFF_FEATS);
  int*      bp    = (int*)(ws + OFF_BP);

  (void)in_sizes; (void)n_in; (void)out_size; (void)ws_size;

  embed_kernel<<<(TT * DP) / 256, 256, 0, stream>>>(words, words1, words2, words3,
                                                    W_emb, emb_h);
  wihT_kernel<<<(2 * DP * G4) / 256, 256, 0, stream>>>(w_ih_f, w_ih_b, wihT);
  whhT_kernel<<<(2 * HID * G4) / 256, 256, 0, stream>>>(w_hh_f, w_hh_b, whhT);
  wtagT_kernel<<<(512 * 16) / 256, 256, 0, stream>>>(W_tag, wtagT);
  xg_gemm_kernel<<<1024, 256, 0, stream>>>(emb_h, wihT, b_ih_f, b_hh_f,
                                           b_ih_b, b_hh_b, xg);
  lstm_rec_kernel<<<2, 1024, 0, stream>>>(xg, whhT, hs_h);
  feats_kernel<<<32, 256, 0, stream>>>(hs_h, wtagT, b_tag, feats);
  viterbi_kernel<<<1, 32, 0, stream>>>(feats, trans, bp, out);
}
